// Model_25125558682000
// MI455X (gfx1250) — matrix operand order repaired and checked by reading, not compiled, not run on hardware
//
#include <hip/hip_runtime.h>
#include <cstdint>

typedef __attribute__((ext_vector_type(16))) _Float16 v16h;
typedef __attribute__((ext_vector_type(4)))  _Float16 v4h;
typedef __attribute__((ext_vector_type(8)))  float    v8f;

#define NA     50000
#define NB     100000
#define NEDGE  300000
#define CDIM   128
#define CE     32
#define NH     4
#define CHD    32
#define NL     2
#define NSEED  512
#define OUTD   8
#define DAY_INV (1.0f / 86400.0f)

// ---------------------------------------------------------------------------
// WMMA GEMM: Y[N,128] = X[N,128] @ W[128,128]   (f32 in, f16 MAC, f32 acc)
// Block = 256 threads = 8 waves; block tile = 64 rows x 128 cols.
// Wave wv owns cols [16*wv, 16*wv+16); loops 4 row-tiles of 16.
// W is staged TRANSPOSED in LDS so both A and B fragments are K-contiguous
// per lane -> compiler merges fragment reads into ds_load_b128 pairs.
// ---------------------------------------------------------------------------
__global__ __launch_bounds__(256)
void k_gemm_wmma(const float* __restrict__ X, const float* __restrict__ W,
                 float* __restrict__ Y, int N)
{
    __shared__ _Float16 WlT[CDIM][CDIM + 8];   // WlT[col][k] = W[k][col]
    __shared__ _Float16 Xl[64][CDIM + 8];      // Xl[row][k]

    const int tid  = threadIdx.x;
    const int wv   = tid >> 5;
    const int lane = tid & 31;
    const int R0   = blockIdx.x * 64;

    // Stage W transposed: coalesced float4 global reads, scattered f16 stores.
    for (int i = tid * 4; i < CDIM * CDIM; i += 256 * 4) {
        const float4 wv4 = *(const float4*)(W + i);
        int k = i >> 7, c = i & 127;
        WlT[c + 0][k] = (_Float16)wv4.x;
        WlT[c + 1][k] = (_Float16)wv4.y;
        WlT[c + 2][k] = (_Float16)wv4.z;
        WlT[c + 3][k] = (_Float16)wv4.w;
    }
    // Stage X tile: float4 reads -> packed 4xf16 LDS stores (row-contiguous).
    for (int i = tid * 4; i < 64 * CDIM; i += 256 * 4) {
        int r = i >> 7, c = i & 127;
        int gr = R0 + r;
        float4 xv;
        if (gr < N) xv = *(const float4*)(X + (size_t)gr * CDIM + c);
        else        { xv.x = 0.f; xv.y = 0.f; xv.z = 0.f; xv.w = 0.f; }
        v4h hx;
        hx[0] = (_Float16)xv.x; hx[1] = (_Float16)xv.y;
        hx[2] = (_Float16)xv.z; hx[3] = (_Float16)xv.w;
        *(v4h*)&Xl[r][c] = hx;
    }
    __syncthreads();

    v8f acc[4];
    #pragma unroll
    for (int t = 0; t < 4; t++)
        #pragma unroll
        for (int i = 0; i < 8; i++) acc[t][i] = 0.0f;

    const int m  = lane & 15;   // row within tile (A) / col within tile (B,C)
    const int hi = lane >> 4;   // half-wave selector

    for (int k0 = 0; k0 < CDIM; k0 += 32) {
        // B fragment 32x16: lane(n=m,hi): col n, half i -> K = k0 + 16*hi + i
        // -> 16 contiguous halfs of WlT[col]
        v16h b;
        #pragma unroll
        for (int i = 0; i < 16; i++)
            b[i] = WlT[wv * 16 + m][k0 + hi * 16 + i];

        #pragma unroll
        for (int t = 0; t < 4; t++) {
            // A fragment 16x32: lane(m,hi): row m,
            // half i -> K = k0 + 16*hi + i  (16 contiguous halfs)
            v16h a;
            #pragma unroll
            for (int i = 0; i < 16; i++) {
                int kk = hi * 16 + i;
                a[i] = Xl[t * 16 + m][k0 + kk];
            }
            acc[t] = __builtin_amdgcn_wmma_f32_16x16x32_f16(
                false, a, false, b, (short)0, acc[t], false, false);
        }
    }

    const int col = wv * 16 + m;
    #pragma unroll
    for (int t = 0; t < 4; t++) {
        #pragma unroll
        for (int v = 0; v < 8; v++) {
            int row = R0 + t * 16 + v + 8 * hi;   // C/D layout: VGPR v -> M = v + 8*hi
            if (row < N) Y[(size_t)row * CDIM + col] = acc[t][v];
        }
    }
}

// ---------------------------------------------------------------------------
// Temporal positional-encoding edge update: eout = ea + PE(rel) @ Wt + bt
// One wave per edge; lane j computes pe_j, then lane k reduces via shuffles.
// ---------------------------------------------------------------------------
__global__ __launch_bounds__(256)
void k_temporal(const float* __restrict__ ea, const float* __restrict__ tt,
                const int* __restrict__ srcI, const int* __restrict__ batch,
                const float* __restrict__ seed_time,
                const float* __restrict__ Wt, const float* __restrict__ bt,
                float* __restrict__ eout)
{
    __shared__ float Wl[CE * CE];
    int tid = threadIdx.x;
    for (int i = tid; i < CE * CE; i += 256) Wl[i] = Wt[i];
    __syncthreads();

    int e    = blockIdx.x * 8 + (tid >> 5);
    int lane = tid & 31;
    if (e >= NEDGE) return;   // wave-uniform

    float rel  = (seed_time[batch[srcI[e]]] - tt[e]) * DAY_INV;
    int   fi   = lane & 15;
    float freq = __expf((float)fi * (1.0f / 15.0f) * (-9.210340371976184f)); // ln(1e-4)
    float v    = rel * freq;
    float p    = (lane < 16) ? sinf(v) : cosf(v);

    float s = 0.0f;
    #pragma unroll
    for (int j = 0; j < 32; j++) {
        float pj = __shfl(p, j, 32);
        s += pj * Wl[j * CE + lane];
    }
    eout[(size_t)e * CE + lane] = ea[(size_t)e * CE + lane] + s + bt[lane];
}

// ---------------------------------------------------------------------------
// Precompute vd[k,h] = sum_ch Wd[k, h*32+ch]*ad[h,ch]  (and ue analogously)
// ---------------------------------------------------------------------------
__global__ void k_prep_vdue(const float* __restrict__ Wdst, const float* __restrict__ adst,
                            const float* __restrict__ Wedge, const float* __restrict__ aedge,
                            float* __restrict__ vd, float* __restrict__ ue)
{
    int lr = blockIdx.x;      // 0..3  = l*2+r
    int k  = threadIdx.x;     // 0..127
    const float* Wd = Wdst + (size_t)lr * CDIM * CDIM;
    const float* ad = adst + (size_t)lr * NH * CHD;
    float* vdo = vd + (size_t)lr * CDIM * NH;
    for (int h = 0; h < NH; h++) {
        float s = 0.0f;
        for (int ch = 0; ch < CHD; ch++) s += Wd[k * CDIM + h * CHD + ch] * ad[h * CHD + ch];
        vdo[k * NH + h] = s;
    }
    if (k < CE) {
        const float* We = Wedge + (size_t)lr * CE * CDIM;
        const float* ae = aedge + (size_t)lr * NH * CHD;
        float* ueo = ue + (size_t)lr * CE * NH;
        for (int h = 0; h < NH; h++) {
            float s = 0.0f;
            for (int ch = 0; ch < CHD; ch++) s += We[k * CDIM + h * CHD + ch] * ae[h * CHD + ch];
            ueo[k * NH + h] = s;
        }
    }
}

// s_src[n,h] = sum_ch hs[n, h*32+ch] * a_s[h,ch]
__global__ void k_ssrc(const float* __restrict__ hs, const float* __restrict__ a_s,
                       float* __restrict__ ssrc, int N)
{
    int i = blockIdx.x * blockDim.x + threadIdx.x;
    if (i >= N * NH) return;
    int n = i / NH, h = i % NH;
    const float* hp = hs + (size_t)n * CDIM + h * CHD;
    const float* ap = a_s + h * CHD;
    float s = 0.0f;
    for (int c = 0; c < CHD; c++) s += hp[c] * ap[c];
    ssrc[i] = s;
}

// s_dst[n,h] = sum_k x[n,k] * vd[k,h]
__global__ void k_sdst(const float* __restrict__ x, const float* __restrict__ vd,
                       float* __restrict__ sdst, int N)
{
    int i = blockIdx.x * blockDim.x + threadIdx.x;
    if (i >= N * NH) return;
    int n = i / NH, h = i % NH;
    const float* xp = x + (size_t)n * CDIM;
    float s = 0.0f;
    for (int k = 0; k < CDIM; k++) s += xp[k] * vd[k * NH + h];
    sdst[i] = s;
}

// s_e[e,h] = sum_k ea[e,k] * ue[k,h]
__global__ void k_se(const float* __restrict__ ea, const float* __restrict__ ue,
                     float* __restrict__ se)
{
    int i = blockIdx.x * blockDim.x + threadIdx.x;
    if (i >= NEDGE * NH) return;
    int e = i / NH, h = i % NH;
    const float* ep = ea + (size_t)e * CE;
    float s = 0.0f;
    for (int k = 0; k < CE; k++) s += ep[k] * ue[k * NH + h];
    se[i] = s;
}

// Monotonic unsigned encoding of float for atomicMax-based segment max
__device__ __forceinline__ unsigned fkey(float f) {
    unsigned b = __float_as_uint(f);
    return (b & 0x80000000u) ? ~b : (b | 0x80000000u);
}
__device__ __forceinline__ float funkey(unsigned u) {
    return (u & 0x80000000u) ? __uint_as_float(u ^ 0x80000000u) : __uint_as_float(~u);
}

__global__ void k_att1(const float* __restrict__ ssrc, const float* __restrict__ sdst,
                       const float* __restrict__ se, const int* __restrict__ srcI,
                       const int* __restrict__ dstI, float* __restrict__ al,
                       unsigned* __restrict__ amax)
{
    int i = blockIdx.x * blockDim.x + threadIdx.x;
    if (i >= NEDGE * NH) return;
    int e = i / NH, h = i % NH;
    float a = ssrc[srcI[e] * NH + h] + sdst[dstI[e] * NH + h] + se[i];
    a = (a > 0.0f) ? a : 0.2f * a;      // leaky_relu 0.2
    al[i] = a;
    atomicMax(amax + dstI[e] * NH + h, fkey(a));
}

__global__ void k_att2(const int* __restrict__ dstI, const unsigned* __restrict__ amax,
                       float* __restrict__ alex, float* __restrict__ den)
{
    int i = blockIdx.x * blockDim.x + threadIdx.x;
    if (i >= NEDGE * NH) return;
    int e = i / NH, h = i % NH;
    float m  = funkey(amax[dstI[e] * NH + h]);
    float ex = expf(alex[i] - m);
    alex[i] = ex;
    atomicAdd(den + dstI[e] * NH + h, ex);
}

__global__ void k_initacc(float* __restrict__ acc, const float* __restrict__ bc, int N)
{
    int i = blockIdx.x * blockDim.x + threadIdx.x;
    if (i < N * CDIM) acc[i] = bc[i & 127];
}

// out[dst] += hs[src] * w ; one wave per edge, 4 cols per lane (head = col/32)
__global__ __launch_bounds__(256)
void k_scatter(const float* __restrict__ hs, const float* __restrict__ alex,
               const float* __restrict__ den, const int* __restrict__ srcI,
               const int* __restrict__ dstI, float* __restrict__ acc)
{
    int e = blockIdx.x * 8 + (threadIdx.x >> 5);
    if (e >= NEDGE) return;   // wave-uniform
    int lane = threadIdx.x & 31;
    int c0 = lane * 4;
    int h  = c0 >> 5;
    int s  = srcI[e], d = dstI[e];
    float w = alex[e * NH + h] / (den[d * NH + h] + 1e-16f);
    const float4 hv = *(const float4*)(hs + (size_t)s * CDIM + c0);
    float* ap = acc + (size_t)d * CDIM + c0;
    atomicAdd(ap + 0, hv.x * w);
    atomicAdd(ap + 1, hv.y * w);
    atomicAdd(ap + 2, hv.z * w);
    atomicAdd(ap + 3, hv.w * w);
}

// LayerNorm + ReLU, one wave per row, shuffle reductions (wave32)
__global__ __launch_bounds__(256)
void k_lnrelu(const float* __restrict__ acc, const float* __restrict__ lw,
              const float* __restrict__ lb, float* __restrict__ xout, int N)
{
    int n = blockIdx.x * 8 + (threadIdx.x >> 5);
    if (n >= N) return;       // wave-uniform
    int lane = threadIdx.x & 31;
    int c0 = lane * 4;
    float4 v = *(const float4*)(acc + (size_t)n * CDIM + c0);
    float s = v.x + v.y + v.z + v.w;
    for (int m = 16; m > 0; m >>= 1) s += __shfl_xor(s, m, 32);
    float mean = s * (1.0f / CDIM);
    float dx = v.x - mean, dy = v.y - mean, dz = v.z - mean, dw = v.w - mean;
    float q = dx * dx + dy * dy + dz * dz + dw * dw;
    for (int m = 16; m > 0; m >>= 1) q += __shfl_xor(q, m, 32);
    float rstd = rsqrtf(q * (1.0f / CDIM) + 1e-5f);
    float4 o;
    o.x = fmaxf(dx * rstd * lw[c0 + 0] + lb[c0 + 0], 0.0f);
    o.y = fmaxf(dy * rstd * lw[c0 + 1] + lb[c0 + 1], 0.0f);
    o.z = fmaxf(dz * rstd * lw[c0 + 2] + lb[c0 + 2], 0.0f);
    o.w = fmaxf(dw * rstd * lw[c0 + 3] + lb[c0 + 3], 0.0f);
    *(float4*)(xout + (size_t)n * CDIM + c0) = o;
}

__global__ void k_head(const float* __restrict__ xA, const float* __restrict__ Wh,
                       const float* __restrict__ bh, float* __restrict__ out)
{
    int i = blockIdx.x * blockDim.x + threadIdx.x;
    if (i >= NSEED * OUTD) return;
    int s = i / OUTD, o = i % OUTD;
    float acc = bh[o];
    const float* xp = xA + (size_t)s * CDIM;
    for (int k = 0; k < CDIM; k++) acc += xp[k] * Wh[k * OUTD + o];
    out[i] = acc;
}

// ---------------------------------------------------------------------------
extern "C" void kernel_launch(void* const* d_in, const int* in_sizes, int n_in,
                              void* d_out, int out_size, void* d_ws, size_t ws_size,
                              hipStream_t stream)
{
    const float* xA    = (const float*)d_in[0];
    const float* xB    = (const float*)d_in[1];
    const float* ea0   = (const float*)d_in[2];
    const float* ea1   = (const float*)d_in[3];
    const float* Wsrc  = (const float*)d_in[4];
    const float* Wdst  = (const float*)d_in[5];
    const float* Wedge = (const float*)d_in[6];
    const float* asrc  = (const float*)d_in[7];
    const float* adst  = (const float*)d_in[8];
    const float* aedge = (const float*)d_in[9];
    const float* bconv = (const float*)d_in[10];
    const float* lnw   = (const float*)d_in[11];
    const float* lnb   = (const float*)d_in[12];
    const float* Wt    = (const float*)d_in[13];
    const float* bt    = (const float*)d_in[14];
    const float* Wh    = (const float*)d_in[15];
    const float* bh    = (const float*)d_in[16];
    const int*   ei0   = (const int*)d_in[17];
    const int*   ei1   = (const int*)d_in[18];
    const float* t0    = (const float*)d_in[19];
    const float* t1    = (const float*)d_in[20];
    const float* seedt = (const float*)d_in[21];
    const int*   batchA= (const int*)d_in[22];
    const int*   batchB= (const int*)d_in[23];

    // workspace carve-up (~298 MB)
    char* w = (char*)d_ws;
    auto alloc = [&](size_t bytes) -> char* {
        char* p = w; w += (bytes + 255) & ~(size_t)255; return p;
    };
    float*    xAbuf  = (float*)alloc((size_t)NA * CDIM * 4);
    float*    xBbuf  = (float*)alloc((size_t)NB * CDIM * 4);
    float*    accA   = (float*)alloc((size_t)NA * CDIM * 4);
    float*    accB   = (float*)alloc((size_t)NB * CDIM * 4);
    float*    hs     = (float*)alloc((size_t)NB * CDIM * 4);
    float*    ea0t   = (float*)alloc((size_t)NEDGE * CE * 4);
    float*    ea1t   = (float*)alloc((size_t)NEDGE * CE * 4);
    float*    ssrc   = (float*)alloc((size_t)NB * NH * 4);
    float*    sdst   = (float*)alloc((size_t)NB * NH * 4);
    float*    se     = (float*)alloc((size_t)NEDGE * NH * 4);
    float*    alex   = (float*)alloc((size_t)NEDGE * NH * 4);
    unsigned* amax   = (unsigned*)alloc((size_t)NB * NH * 4);
    float*    den    = (float*)alloc((size_t)NB * NH * 4);
    float*    vd_buf = (float*)alloc((size_t)NL * 2 * CDIM * NH * 4);
    float*    ue_buf = (float*)alloc((size_t)NL * 2 * CE * NH * 4);
    (void)ws_size; (void)in_sizes; (void)n_in; (void)out_size;

    k_prep_vdue<<<NL * 2, CDIM, 0, stream>>>(Wdst, adst, Wedge, aedge, vd_buf, ue_buf);
    k_temporal<<<NEDGE / 8, 256, 0, stream>>>(ea0, t0, ei0, batchA, seedt,
                                              Wt, bt, ea0t);
    k_temporal<<<NEDGE / 8, 256, 0, stream>>>(ea1, t1, ei1, batchB, seedt,
                                              Wt + CE * CE, bt + CE, ea1t);

    auto run_gat = [&](const float* x_src, const float* x_dst, int Nsrc, int Ndst,
                       const float* ea_t, const int* ei, int l, int r, float* acc_out) {
        int lr = l * 2 + r;
        const float* Ws = Wsrc  + (size_t)lr * CDIM * CDIM;
        const float* as = asrc  + (size_t)lr * NH * CHD;
        const float* bc = bconv + (size_t)lr * CDIM;
        const float* vd = vd_buf + (size_t)lr * CDIM * NH;
        const float* ue = ue_buf + (size_t)lr * CE * NH;
        const int* srcI = ei;
        const int* dstI = ei + NEDGE;
        int eh = NEDGE * NH;

        k_gemm_wmma<<<(Nsrc + 63) / 64, 256, 0, stream>>>(x_src, Ws, hs, Nsrc);
        k_ssrc<<<(Nsrc * NH + 255) / 256, 256, 0, stream>>>(hs, as, ssrc, Nsrc);
        k_sdst<<<(Ndst * NH + 255) / 256, 256, 0, stream>>>(x_dst, vd, sdst, Ndst);
        k_se<<<(eh + 255) / 256, 256, 0, stream>>>(ea_t, ue, se);
        hipMemsetAsync(amax, 0, (size_t)Ndst * NH * 4, stream);
        hipMemsetAsync(den, 0, (size_t)Ndst * NH * 4, stream);
        k_att1<<<(eh + 255) / 256, 256, 0, stream>>>(ssrc, sdst, se, srcI, dstI, alex, amax);
        k_att2<<<(eh + 255) / 256, 256, 0, stream>>>(dstI, amax, alex, den);
        k_initacc<<<((size_t)Ndst * CDIM + 255) / 256, 256, 0, stream>>>(acc_out, bc, Ndst);
        k_scatter<<<(NEDGE + 7) / 8, 256, 0, stream>>>(hs, alex, den, srcI, dstI, acc_out);
    };

    const float* curA = xA;
    const float* curB = xB;
    for (int l = 0; l < NL; l++) {
        run_gat(curA, curB, NA, NB, ea0t, ei0, l, 0, accB);   // A -> B  (rel 0)
        run_gat(curB, curA, NB, NA, ea1t, ei1, l, 1, accA);   // B -> A  (rel 1)
        k_lnrelu<<<(NA + 7) / 8, 256, 0, stream>>>(
            accA, lnw + (size_t)(l * 2 + 0) * CDIM, lnb + (size_t)(l * 2 + 0) * CDIM, xAbuf, NA);
        k_lnrelu<<<(NB + 7) / 8, 256, 0, stream>>>(
            accB, lnw + (size_t)(l * 2 + 1) * CDIM, lnb + (size_t)(l * 2 + 1) * CDIM, xBbuf, NB);
        curA = xAbuf;
        curB = xBbuf;
    }
    k_head<<<(NSEED * OUTD + 255) / 256, 256, 0, stream>>>(curA, Wh, bh, (float*)d_out);
}
